// GraphAttentionLayer_12575664243352
// MI455X (gfx1250) — matrix operand order repaired and checked by reading, not compiled, not run on hardware
//
#include <hip/hip_runtime.h>

#define N_NODES 2048
#define F_IN    512
#define F_OUT   256
#define BATCH   8
#define NEG_BIG (-9000000000000000.0f)
#define LR_ALPHA 0.2f
#define JB 64        // j-block width (two K=32 WMMA sweeps)
#define PST 68       // padded LDS row stride for the P tile (floats)

typedef __bf16 bf16_t;
typedef __attribute__((ext_vector_type(8)))  __bf16 v8bf;
typedef __attribute__((ext_vector_type(16))) __bf16 v16bf;
typedef __attribute__((ext_vector_type(8)))  float  v8f;

static __device__ inline v16bf cat16(v8bf lo, v8bf hi) {
  return __builtin_shufflevector(lo, hi, 0,1,2,3,4,5,6,7,8,9,10,11,12,13,14,15);
}

static __device__ inline v8bf cvt8(const float* __restrict__ p) {
  v8bf r;
  #pragma unroll
  for (int i = 0; i < 8; ++i) r[i] = (__bf16)p[i];
  return r;
}

// ---------------------------------------------------------------------------
// Prep 0: x (fp32) -> xb (bf16), contiguous
// ---------------------------------------------------------------------------
__global__ void cvt_x_bf16(const float* __restrict__ x, bf16_t* __restrict__ xb,
                           int n) {
  int i = blockIdx.x * blockDim.x + threadIdx.x;
  if (i < n) xb[i] = (__bf16)x[i];
}

// ---------------------------------------------------------------------------
// Prep 1: W (F_IN x F_OUT fp32) -> Wtb (F_OUT x F_IN bf16), transposed
// ---------------------------------------------------------------------------
__global__ void cvt_w_t_bf16(const float* __restrict__ W, bf16_t* __restrict__ Wtb) {
  int i = blockIdx.x * blockDim.x + threadIdx.x;   // over F_OUT*F_IN
  if (i >= F_OUT * F_IN) return;
  int n = i / F_IN, k = i % F_IN;
  Wtb[i] = (__bf16)W[(size_t)k * F_OUT + n];
}

// ---------------------------------------------------------------------------
// Kernel 1: h[b,i,n] = sum_k x[b,i,k] * W[k,n]  via bf16 WMMA (K=32/issue).
// One wave per 16x16 tile. Writes h fp32 (for f,g) and h^T bf16 (attention B).
// ---------------------------------------------------------------------------
__global__ __launch_bounds__(32)
void gat_gemm_h(const bf16_t* __restrict__ xb, const bf16_t* __restrict__ Wtb,
                float* __restrict__ h, bf16_t* __restrict__ hbt) {
  const int lane = threadIdx.x;          // 0..31 (wave32)
  const int n0   = blockIdx.x * 16;
  const int i0   = blockIdx.y * 16;
  const int b    = blockIdx.z;
  const int m    = lane & 15;            // A row / B col within tile
  const int hi   = lane >> 4;            // lane half selects K sub-chunk

  const bf16_t* xrow = xb  + ((size_t)b * N_NODES + i0 + m) * F_IN;
  const bf16_t* wrow = Wtb + (size_t)(n0 + m) * F_IN;

  v8f acc = {};
  for (int k = 0; k < F_IN; k += 32) {
    // A 16x32 bf16: V0-3 = K(8*hi..8*hi+7), V4-7 = K(16+8*hi..+7)
    v16bf A = cat16(*(const v8bf*)(xrow + k + 8 * hi),
                    *(const v8bf*)(xrow + k + 16 + 8 * hi));
    // B 32x16 bf16: V0-3 = K(8*hi..8*hi+7), V4-7 = K(16+8*hi..+7), col n striped across lanes
    v16bf B = cat16(*(const v8bf*)(wrow + k + 8 * hi),
                    *(const v8bf*)(wrow + k + 16 + 8 * hi));
    acc = __builtin_amdgcn_wmma_f32_16x16x32_bf16(
        false, A, false, B, (short)0, acc, false, false);
  }

  // C/D: lane holds col n0+m; VGPR v -> row i0 + rbase + v
  const int rbase = hi * 8;
  float* hout = h + ((size_t)b * N_NODES + i0) * F_OUT + n0 + m;
  v8bf pk;
  #pragma unroll
  for (int v = 0; v < 8; ++v) {
    hout[(size_t)(rbase + v) * F_OUT] = acc[v];
    pk[v] = (__bf16)acc[v];
  }
  // h^T bf16: row (n0+m), cols i0+rbase..+7 contiguous -> one 16B store
  *(v8bf*)(hbt + ((size_t)b * F_OUT + n0 + m) * N_NODES + i0 + rbase) = pk;
}

// ---------------------------------------------------------------------------
// Kernel 2: f = h @ a[:256], g = h @ a[256:]   (one thread per (b,n) row)
// ---------------------------------------------------------------------------
__global__ void gat_fg(const float* __restrict__ h, const float* __restrict__ a,
                       float* __restrict__ f, float* __restrict__ g) {
  int idx = blockIdx.x * blockDim.x + threadIdx.x;   // b*N + n
  if (idx >= BATCH * N_NODES) return;
  const float* hr = h + (size_t)idx * F_OUT;
  float fs = 0.f, gs = 0.f;
  for (int o = 0; o < F_OUT; ++o) {
    float hv = hr[o];
    fs += hv * a[o];
    gs += hv * a[F_OUT + o];
  }
  f[idx] = fs;
  g[idx] = gs;
}

// ---------------------------------------------------------------------------
// Kernel 3: fused masked-softmax attention + (att @ h) + ELU.
// One block (16 waves) per (batch, 16-row block). Wave w owns output cols
// [16w,16w+16). Online softmax fp32 in LDS; P@H in bf16 WMMA (fp32 accum).
// ---------------------------------------------------------------------------
__global__ __launch_bounds__(512)
void gat_attn(const bf16_t* __restrict__ hbt, const int* __restrict__ adj,
              const float* __restrict__ f, const float* __restrict__ g,
              float* __restrict__ out) {
  __shared__ float p_sh[16 * PST];   // e / p tile, 16 rows x JB cols (padded)
  __shared__ float f_sh[16];
  __shared__ float m_sh[16];         // running row max
  __shared__ float l_sh[16];         // running row sum
  __shared__ float s_sh[16];         // per-block rescale exp(m_old - m_new)

  const int tid  = threadIdx.x;      // 0..511
  const int lane = tid & 31;
  const int wave = tid >> 5;         // 0..15
  const int i0   = blockIdx.x * 16;
  const int b    = blockIdx.y;

  if (tid < 16) {
    f_sh[tid] = f[(size_t)b * N_NODES + i0 + tid];
    m_sh[tid] = NEG_BIG;
    l_sh[tid] = 0.f;
  }
  __syncthreads();

  const int n0    = wave * 16;
  const int cn    = lane & 15;       // A row (= P row) and output col in tile
  const int hi    = lane >> 4;
  const int rbase = hi * 8;

  const int*    adjrow = adj + ((size_t)b * N_NODES + i0) * N_NODES;
  const float*  gb     = g + (size_t)b * N_NODES;
  const bf16_t* hrow   = hbt + ((size_t)b * F_OUT + n0 + cn) * N_NODES;

  v8f acc = {};

  for (int jb = 0; jb < N_NODES; jb += JB) {
    // ---- e tile: leaky_relu(f_i + g_j) masked by adj; 2 elems per thread ----
    #pragma unroll
    for (int t = 0; t < 2; ++t) {
      int idx = tid + t * 512;
      int r = idx >> 6, c = idx & 63;
      float z = f_sh[r] + gb[jb + c];
      z = z > 0.f ? z : LR_ALPHA * z;
      int av = adjrow[(size_t)r * N_NODES + jb + c];
      p_sh[r * PST + c] = (av > 0) ? z : NEG_BIG;
      if (jb + JB < N_NODES)   // pull next adj block toward the WGP
        __builtin_prefetch(&adjrow[(size_t)r * N_NODES + jb + JB + c], 0, 0);
    }
    __syncthreads();

    // ---- online max update (16 lanes of wave 0) ----
    if (tid < 16) {
      float bm = NEG_BIG;
      for (int c = 0; c < JB; ++c) bm = fmaxf(bm, p_sh[tid * PST + c]);
      float mold = m_sh[tid];
      float mnew = fmaxf(mold, bm);
      s_sh[tid] = __expf(mold - mnew);
      m_sh[tid] = mnew;
    }
    __syncthreads();

    // ---- p = exp(e - m_new) ----
    #pragma unroll
    for (int t = 0; t < 2; ++t) {
      int idx = tid + t * 512;
      int r = idx >> 6, c = idx & 63;
      p_sh[r * PST + c] = __expf(p_sh[r * PST + c] - m_sh[r]);
    }
    __syncthreads();

    // ---- row-sum update (overlaps with WMMA below) ----
    if (tid < 16) {
      float s = 0.f;
      for (int c = 0; c < JB; ++c) s += p_sh[tid * PST + c];
      l_sh[tid] = l_sh[tid] * s_sh[tid] + s;
    }

    // ---- rescale accumulator rows by exp(m_old - m_new) ----
    #pragma unroll
    for (int v = 0; v < 8; ++v) acc[v] *= s_sh[rbase + v];

    // ---- acc += P(16xJB) @ H(JBx16): bf16 WMMA, K=32 per issue ----
    #pragma unroll
    for (int k0 = 0; k0 < JB; k0 += 32) {
      const float* pr = &p_sh[cn * PST + k0];
      v16bf A = cat16(cvt8(pr + 8 * hi), cvt8(pr + 16 + 8 * hi));
      const bf16_t* hp = hrow + jb + k0 + 8 * hi;
      v16bf B = cat16(*(const v8bf*)hp, *(const v8bf*)(hp + 16));
      acc = __builtin_amdgcn_wmma_f32_16x16x32_bf16(
          false, A, false, B, (short)0, acc, false, false);
    }
    __syncthreads();
  }

  // ---- finalize: divide by l, ELU, store ----
  float* outp = out + ((size_t)b * N_NODES + i0) * F_OUT + n0 + cn;
  #pragma unroll
  for (int v = 0; v < 8; ++v) {
    float val = acc[v] / l_sh[rbase + v];
    val = val > 0.f ? val : (__expf(val) - 1.f);
    outp[(size_t)(rbase + v) * F_OUT] = val;
  }
}

// ---------------------------------------------------------------------------
extern "C" void kernel_launch(void* const* d_in, const int* in_sizes, int n_in,
                              void* d_out, int out_size, void* d_ws, size_t ws_size,
                              hipStream_t stream) {
  const float* x   = (const float*)d_in[0];
  const int*   adj = (const int*)  d_in[1];
  const float* W   = (const float*)d_in[2];
  const float* a   = (const float*)d_in[3];
  float* out = (float*)d_out;

  // workspace layout (all chunks 256B-aligned by construction)
  char* w = (char*)d_ws;
  float*  h   = (float*)w;                       w += (size_t)BATCH * N_NODES * F_OUT * 4;
  float*  f   = (float*)w;                       w += (size_t)BATCH * N_NODES * 4;
  float*  g   = (float*)w;                       w += (size_t)BATCH * N_NODES * 4;
  bf16_t* xb  = (bf16_t*)w;                      w += (size_t)BATCH * N_NODES * F_IN * 2;
  bf16_t* Wtb = (bf16_t*)w;                      w += (size_t)F_OUT * F_IN * 2;
  bf16_t* hbt = (bf16_t*)w;                      // BATCH * F_OUT * N_NODES bf16

  const int nx = BATCH * N_NODES * F_IN;
  cvt_x_bf16<<<(nx + 255) / 256, 256, 0, stream>>>(x, xb, nx);
  cvt_w_t_bf16<<<(F_OUT * F_IN + 255) / 256, 256, 0, stream>>>(W, Wtb);

  dim3 g1(F_OUT / 16, N_NODES / 16, BATCH);
  gat_gemm_h<<<g1, 32, 0, stream>>>(xb, Wtb, h, hbt);

  gat_fg<<<(BATCH * N_NODES + 255) / 256, 256, 0, stream>>>(h, a, f, g);

  dim3 g2(N_NODES / 16, BATCH);
  gat_attn<<<g2, 512, 0, stream>>>(hbt, adj, f, g, out);
}
